// DecoderModelWithAttention_43319040147679
// MI455X (gfx1250) — matrix operand order repaired and checked by reading, not compiled, not run on hardware
//
#include <hip/hip_runtime.h>
#include <hip/hip_bf16.h>
#include <math.h>

// ---- problem constants ----
#define B_ 256
#define S_ 24
#define N_ 207
#define U_ 64
#define A_ 256
#define H_ 13248            // N_*U_
#define KT_ATTN 414         // H_/32
#define KPAD_ 224           // adj K padded (7*32)
#define MPAD_ 208           // adj M padded (13*16)
#define FPAD_ 144           // feature dim padded (9*16), logical 129

typedef __attribute__((ext_vector_type(16))) __bf16 v16bf;
typedef __attribute__((ext_vector_type(8)))  float  v8f;

// ---------- CDNA5 async global->LDS copy (ASYNCcnt-tracked) ----------
__device__ inline unsigned lds_off(const void* p) {
  // shared-aperture flat address: addr[31:0] is the LDS offset (ISA 10.2)
  return (unsigned)(unsigned long long)p;
}
__device__ inline void async_b128(unsigned lds_dst, const void* gsrc) {
  asm volatile("global_load_async_to_lds_b128 %0, %1, off"
               :: "v"(lds_dst), "v"(gsrc) : "memory");
}
__device__ inline void wait_async0() {
  asm volatile("s_wait_asynccnt 0x0" ::: "memory");
}

// ---------- WMMA fragment helpers ----------
// A-matrix (16x32 bf16) lane layout: lanes 0-15 -> M=lane, K-groups {0..7,16..23};
// lanes 16-31 -> M=lane-16, K-groups {8..15,24..31}.
__device__ inline v16bf load_a_frag_f32(const float* __restrict__ rowPtr, int kBase) {
  const int L = threadIdx.x & 31;
  const int g = (L & 16) ? 8 : 0;
  const float* p0 = rowPtr + kBase + g;
  const float* p1 = rowPtr + kBase + 16 + g;
  v16bf a;
#pragma unroll
  for (int i = 0; i < 8; ++i) a[i] = (__bf16)p0[i];
#pragma unroll
  for (int i = 0; i < 8; ++i) a[8 + i] = (__bf16)p1[i];
  return a;
}

// C/D (16x16 f32): VGPR i -> M = i + (lane>=16 ? 8 : 0), N = lane&15.
__device__ inline void store_acc(float* __restrict__ C, int ld, int rowBase, int colBase, v8f acc) {
  const int L = threadIdx.x & 31;
  const int mOff = (L & 16) ? 8 : 0;
  const int n = colBase + (L & 15);
#pragma unroll
  for (int i = 0; i < 8; ++i) C[(size_t)(rowBase + mOff + i) * ld + n] = acc[i];
}

__device__ inline v8f vzero8() { v8f v = {0.f,0.f,0.f,0.f,0.f,0.f,0.f,0.f}; return v; }

// ---------- 1. pack We (fp32 -> bf16) into B-fragment layout ----------
// B-frag element i at lane L of tile (kt,ct): k = kt*32 + ((L&16)?8:0) + (i<8 ? i : 16+(i-8)), n = ct*16 + (L&15)
__global__ void cvt_we_kernel(const float* __restrict__ attn_W, __bf16* __restrict__ Wp) {
  int t = blockIdx.x * blockDim.x + threadIdx.x;
  if (t >= H_ * A_) return;
  int i  = t & 15;
  int L  = (t >> 4) & 31;
  int ct = (t >> 9) & 15;
  int kt = t >> 13;
  int k = kt * 32 + ((L & 16) ? 8 : 0) + (i < 8 ? i : 16 + (i - 8));
  int a = ct * 16 + (L & 15);
  Wp[t] = (__bf16)attn_W[(size_t)(H_ + k) * A_ + a];
}

// ---------- 2. pack adj (zero-padded 208x224) into A-fragment layout ----------
__global__ void cvt_adj_kernel(const float* __restrict__ adj, __bf16* __restrict__ adjA) {
  int t = blockIdx.x * blockDim.x + threadIdx.x;
  if (t >= MPAD_ * KPAD_) return;
  int i  = t & 15;
  int L  = (t >> 4) & 31;
  int r  = t >> 9;
  int kt = r % 7;
  int rt = r / 7;
  int n = rt * 16 + (L & 15);
  int k = kt * 32 + ((L & 16) ? 16 : 0) + i;
  float v = (n < N_ && k < N_) ? adj[n * N_ + k] : 0.0f;
  adjA[t] = (__bf16)v;
}

// ---------- 3. attention GEMM with async double-buffered LDS staging of B ----------
// rows [enc(6144); hid(256)] x We -> attn_out.  Block = 4 waves = 4 row tiles,
// 128 cols (8 tiles); all waves share the same B tiles via LDS.
__global__ __launch_bounds__(128) void attn_gemm_kernel(const float* __restrict__ enc0,
                                                        const float* __restrict__ hid0,
                                                        const __bf16* __restrict__ Wp,
                                                        float* __restrict__ out) {
  __shared__ __align__(32) char sb[2][8192];          // 8 tiles * 1KB, double buffered
  const int rowTile = blockIdx.x * 4 + (threadIdx.x >> 5);   // 0..399
  const int ctBase  = blockIdx.y * 8;                        // 0 or 8 (128 cols)
  const int L = threadIdx.x & 31;
  const int r = rowTile * 16 + (L & 15);
  const float* aRow;
  if (r < B_ * S_) {
    int bb = r / S_;
    int s  = r - bb * S_;
    aRow = enc0 + ((size_t)s * B_ + bb) * H_;     // enc[b][s][:] = encoder_outputs[s][b][:]
  } else {
    aRow = hid0 + (size_t)(r - B_ * S_) * H_;
  }
  const char* Wbase = (const char*)Wp;
  const unsigned sb0 = lds_off(&sb[0][0]);

  // prologue: async-copy B tiles for kt=0 into buffer 0 (128 lanes * 16B * 4 = 8KB)
  {
    const char* src = Wbase + (size_t)ctBase * 1024 + threadIdx.x * 16;
#pragma unroll
    for (int j = 0; j < 4; ++j)
      async_b128(sb0 + threadIdx.x * 16 + j * 2048, src + j * 2048);
  }

  v8f acc[8];
#pragma unroll
  for (int c = 0; c < 8; ++c) acc[c] = vzero8();

  for (int kt = 0; kt < KT_ATTN; ++kt) {
    const int cur = kt & 1;
    wait_async0();        // own copies for kt landed
    __syncthreads();      // everyone's copies landed; everyone done reading other buffer
    if (kt + 1 < KT_ATTN) {
      const char* src = Wbase + ((size_t)(kt + 1) * 16 + ctBase) * 1024 + threadIdx.x * 16;
      unsigned dst = sb0 + (unsigned)((cur ^ 1) * 8192) + threadIdx.x * 16;
#pragma unroll
      for (int j = 0; j < 4; ++j)
        async_b128(dst + j * 2048, src + j * 2048);
    }
    v16bf a = load_a_frag_f32(aRow, kt * 32);
    const char* bbuf = &sb[cur][0];
#pragma unroll
    for (int c = 0; c < 8; ++c) {
      v16bf bf = *(const v16bf*)(bbuf + (c * 32 + L) * 32);
      acc[c] = __builtin_amdgcn_wmma_f32_16x16x32_bf16(false, a, false, bf,
                                                       (short)0, acc[c], false, false);
    }
  }
#pragma unroll
  for (int c = 0; c < 8; ++c)
    store_acc(out, A_, rowTile * 16, (ctBase + c) * 16, acc[c]);
}

// ---------- 4. scores + softmax over S ----------
__global__ __launch_bounds__(256) void scores_softmax_kernel(const float* __restrict__ attn_out,
                                                             const float* __restrict__ attn_b,
                                                             const float* __restrict__ attn_v,
                                                             float* __restrict__ a_sm) {
  __shared__ float red[256];
  __shared__ float sc[S_];
  const int b = blockIdx.x;
  const int t = threadIdx.x;
  const float bias = attn_b[t];
  const float vv   = attn_v[t];
  const float hw   = attn_out[(size_t)(B_ * S_ + b) * A_ + t];   // hid @ Wh
  for (int s = 0; s < S_; ++s) {
    float e = attn_out[(size_t)(b * S_ + s) * A_ + t] + hw + bias;
    red[t] = tanhf(e) * vv;
    __syncthreads();
    for (int off = 128; off > 0; off >>= 1) {
      if (t < off) red[t] += red[t + off];
      __syncthreads();
    }
    if (t == 0) sc[s] = red[0];
    __syncthreads();
  }
  if (t == 0) {
    float mx = sc[0];
    for (int s = 1; s < S_; ++s) mx = fmaxf(mx, sc[s]);
    float sum = 0.f;
    for (int s = 0; s < S_; ++s) { sc[s] = expf(sc[s] - mx); sum += sc[s]; }
    float inv = 1.f / sum;
    for (int s = 0; s < S_; ++s) a_sm[b * S_ + s] = sc[s] * inv;
  }
}

// ---------- 5. weighted context: weighted[b,h] = sum_s a[b,s]*enc[b,s,h] ----------
__global__ void weighted_kernel(const float* __restrict__ a_sm, const float* __restrict__ enc0,
                                float* __restrict__ weighted) {
  int t = blockIdx.x * blockDim.x + threadIdx.x;
  if (t >= B_ * H_) return;
  int b = t / H_;
  int h = t - b * H_;
  const float* ap = a_sm + b * S_;
  float acc = 0.f;
#pragma unroll 4
  for (int s = 0; s < S_; ++s)
    acc = fmaf(ap[s], enc0[((size_t)s * B_ + b) * H_ + h], acc);
  weighted[t] = acc;
}

// ---------- 6. build F = [inputs | weighted | h] bf16 in packed B-fragment layout ----------
// packed index: (((b*7 + kt)*9 + ct)*32 + L)*16 + i ; m = kt*32+((L&16)?16:0)+i ; f = ct*16+(L&15)
__global__ void buildF_kernel(const float* __restrict__ inputs, const float* __restrict__ weighted,
                              const float* __restrict__ hid0, __bf16* __restrict__ F) {
  int t = blockIdx.x * blockDim.x + threadIdx.x;
  if (t >= B_ * 7 * 9 * 32 * 16) return;
  int i  = t & 15;
  int L  = (t >> 4) & 31;
  int r  = t >> 9;
  int ct = r % 9;  r /= 9;
  int kt = r % 7;
  int b  = r / 7;
  int m = kt * 32 + ((L & 16) ? 16 : 0) + i;
  int f = ct * 16 + (L & 15);
  float val = 0.f;
  if (m < N_ && f < 129) {
    if (f == 0)       val = inputs[b * N_ + m];
    else if (f <= 64) val = weighted[(size_t)b * H_ + m * U_ + (f - 1)];
    else              val = hid0[(size_t)b * H_ + m * U_ + (f - 65)];
  }
  F[t] = (__bf16)val;
}

// ---------- 7/9. adj GEMM: M[b] = adj(208x224) @ F[b](224x144) ----------
// One block per batch element, 13 waves = 13 row tiles; F[b] staged once into
// LDS via async copies (double buffered), shared by all 13 waves.
__global__ __launch_bounds__(416) void adj_gemm_kernel(const v16bf* __restrict__ adjA,
                                                       const __bf16* __restrict__ Fp,
                                                       float* __restrict__ Mout) {
  __shared__ __align__(32) char sf[2][9216];          // 9 tiles * 1KB, double buffered
  const int b   = blockIdx.x;
  const int wid = threadIdx.x >> 5;      // 0..12 = row tile
  const int L   = threadIdx.x & 31;
  const char* Fb = (const char*)Fp + (size_t)b * 7 * 9 * 1024;
  const unsigned sf0 = lds_off(&sf[0][0]);

  // prologue: copy kt=0 tiles (9KB): threads 0..287 copy 32B each
  if (threadIdx.x < 288) {
    unsigned dst = sf0 + threadIdx.x * 32;
    const char* src = Fb + threadIdx.x * 32;
    async_b128(dst, src);
    async_b128(dst + 16, src + 16);
  }

  v8f acc[9];
#pragma unroll
  for (int c = 0; c < 9; ++c) acc[c] = vzero8();

  for (int kt = 0; kt < 7; ++kt) {
    const int cur = kt & 1;
    wait_async0();
    __syncthreads();
    if (kt + 1 < 7 && threadIdx.x < 288) {
      unsigned dst = sf0 + (unsigned)((cur ^ 1) * 9216) + threadIdx.x * 32;
      const char* src = Fb + (size_t)(kt + 1) * 9216 + threadIdx.x * 32;
      async_b128(dst, src);
      async_b128(dst + 16, src + 16);
    }
    v16bf a = adjA[((size_t)wid * 7 + kt) * 32 + L];
    const char* bbuf = &sf[cur][0];
#pragma unroll
    for (int c = 0; c < 9; ++c) {
      v16bf bf = *(const v16bf*)(bbuf + (c * 32 + L) * 32);
      acc[c] = __builtin_amdgcn_wmma_f32_16x16x32_bf16(false, a, false, bf,
                                                       (short)0, acc[c], false, false);
    }
  }
  float* Mb = Mout + (size_t)b * MPAD_ * FPAD_;
#pragma unroll
  for (int c = 0; c < 9; ++c) store_acc(Mb, FPAD_, wid * 16, c * 16, acc[c]);
}

// ---------- 8. gates: ru = sigmoid(M @ W_ru + b_ru); store u, write bf16(r*h) into packed F ----------
__global__ void ru_kernel(const float* __restrict__ M, const float* __restrict__ W_ru,
                          const float* __restrict__ b_ru, const float* __restrict__ hid0,
                          float* __restrict__ u_gate, __bf16* __restrict__ F) {
  int t = blockIdx.x * blockDim.x + threadIdx.x;
  if (t >= B_ * N_ * 128) return;
  int j   = t & 127;
  int idx = t >> 7;
  int n = idx % N_;
  int b = idx / N_;
  const float* Mrow = M + ((size_t)b * MPAD_ + n) * FPAD_;
  float acc = b_ru[j];
  for (int f = 0; f < 129; ++f) acc = fmaf(Mrow[f], W_ru[f * 128 + j], acc);
  float g = 1.0f / (1.0f + expf(-acc));
  if (j < 64) {
    // r-gate: overwrite F's h slot (feature 65+j, node m=n) in packed layout
    float h = hid0[(size_t)b * H_ + n * U_ + j];
    int f = 65 + j;
    int kt = n >> 5;
    int ct = f >> 4;
    int L  = (n & 16) | (f & 15);
    int i  = n & 15;
    F[((((size_t)b * 7 + kt) * 9 + ct) * 32 + L) * 16 + i] = (__bf16)(g * h);
  } else {
    u_gate[((size_t)b * N_ + n) * U_ + (j - 64)] = g;
  }
}

// ---------- 10. candidate + new_h + projection ----------
__global__ __launch_bounds__(64) void final_kernel(const float* __restrict__ M,
                                                   const float* __restrict__ W_c,
                                                   const float* __restrict__ b_c,
                                                   const float* __restrict__ hid0,
                                                   const float* __restrict__ u_gate,
                                                   const float* __restrict__ weighted,
                                                   const float* __restrict__ inputs,
                                                   const float* __restrict__ proj_W,
                                                   const float* __restrict__ proj_b,
                                                   float* __restrict__ out) {
  __shared__ float red[64];
  const int bid = blockIdx.x;
  const int b = bid / N_;
  const int n = bid - b * N_;
  const int j = threadIdx.x;
  const float* Mrow = M + ((size_t)b * MPAD_ + n) * FPAD_;
  float acc = b_c[j];
  for (int f = 0; f < 129; ++f) acc = fmaf(Mrow[f], W_c[f * U_ + j], acc);
  float c  = tanhf(acc);
  float u  = u_gate[((size_t)b * N_ + n) * U_ + j];
  float h  = hid0[(size_t)b * H_ + n * U_ + j];
  float nh = u * h + (1.0f - u) * c;
  out[B_ * N_ + (size_t)b * H_ + n * U_ + j] = nh;          // new_h output
  float w = weighted[(size_t)b * H_ + n * U_ + j];
  red[j] = nh * proj_W[j] + w * proj_W[U_ + j];
  __syncthreads();
  for (int off = 32; off > 0; off >>= 1) {
    if (j < off) red[j] += red[j + off];
    __syncthreads();
  }
  if (j == 0)
    out[(size_t)b * N_ + n] = red[0] + inputs[b * N_ + n] * proj_W[128] + proj_b[0];
}

// ---------- host launcher ----------
extern "C" void kernel_launch(void* const* d_in, const int* in_sizes, int n_in,
                              void* d_out, int out_size, void* d_ws, size_t ws_size,
                              hipStream_t stream) {
  (void)in_sizes; (void)n_in; (void)out_size; (void)ws_size;
  const float* inputs = (const float*)d_in[0];
  const float* enc0   = (const float*)d_in[1];
  const float* hid0   = (const float*)d_in[2];
  const float* adj    = (const float*)d_in[3];
  const float* attn_W = (const float*)d_in[4];
  const float* attn_b = (const float*)d_in[5];
  const float* attn_v = (const float*)d_in[6];
  const float* W_ru   = (const float*)d_in[7];
  const float* b_ru   = (const float*)d_in[8];
  const float* W_c    = (const float*)d_in[9];
  const float* b_c    = (const float*)d_in[10];
  const float* proj_W = (const float*)d_in[11];
  const float* proj_b = (const float*)d_in[12];
  float* out = (float*)d_out;

  // workspace layout (bytes, all 256-aligned)
  char* ws = (char*)d_ws;
  float*  attn_out = (float*)(ws + 0);              // 6400*256*4      = 6,553,600
  __bf16* Wp       = (__bf16*)(ws + 6553600);       // 13248*256*2     = 6,782,976
  float*  a_sm     = (float*)(ws + 13336576);       // 256*24*4        = 24,576
  float*  weighted = (float*)(ws + 13361152);       // 256*13248*4     = 13,565,952
  __bf16* adjA     = (__bf16*)(ws + 26927104);      // 208*224*2       = 93,184
  __bf16* Fp       = (__bf16*)(ws + 27020288);      // 256*224*144*2   = 16,515,072
  float*  Mbuf     = (float*)(ws + 43535360);       // 256*208*144*4   = 30,670,848
  float*  u_gate   = (float*)(ws + 74206208);       // 256*207*64*4    = 13,565,952
  // total ~87.8 MB

  // 1. pack We -> bf16 fragments
  cvt_we_kernel<<<(H_ * A_ + 255) / 256, 256, 0, stream>>>(attn_W, Wp);
  // 2. pack adj -> bf16 A-fragments (padded)
  cvt_adj_kernel<<<(MPAD_ * KPAD_ + 255) / 256, 256, 0, stream>>>(adj, adjA);
  // 3. big attention GEMM (WMMA bf16 + async LDS staging)
  attn_gemm_kernel<<<dim3(100, 2), 128, 0, stream>>>(enc0, hid0, Wp, attn_out);
  // 4. scores + softmax
  scores_softmax_kernel<<<B_, 256, 0, stream>>>(attn_out, attn_b, attn_v, a_sm);
  // 5. weighted context
  weighted_kernel<<<(B_ * H_ + 255) / 256, 256, 0, stream>>>(a_sm, enc0, weighted);
  // 6. build F = [x | h] packed bf16
  buildF_kernel<<<(B_ * 7 * 9 * 32 * 16 + 255) / 256, 256, 0, stream>>>(inputs, weighted, hid0, Fp);
  // 7. m = adj @ [x|h]  (WMMA bf16 + async LDS staging)
  adj_gemm_kernel<<<B_, 416, 0, stream>>>((const v16bf*)adjA, Fp, Mbuf);
  // 8. gates; rewrite F's h-slot with r*h
  ru_kernel<<<(B_ * N_ * 128 + 255) / 256, 256, 0, stream>>>(Mbuf, W_ru, b_ru, hid0, u_gate, Fp);
  // 9. mc = adj @ [x | r*h]  (WMMA bf16, overwrites Mbuf)
  adj_gemm_kernel<<<B_, 416, 0, stream>>>((const v16bf*)adjA, Fp, Mbuf);
  // 10. candidate, new_h, projection
  final_kernel<<<B_ * N_, 64, 0, stream>>>(Mbuf, W_c, b_c, hid0, u_gate, weighted,
                                           inputs, proj_W, proj_b, out);
}
